// TestModule_1443109012263
// MI455X (gfx1250) — matrix operand order repaired and checked by reading, not compiled, not run on hardware
//
#include <hip/hip_runtime.h>

typedef __attribute__((ext_vector_type(16))) _Float16 v16h;
typedef __attribute__((ext_vector_type(8)))  _Float16 v8h;
typedef __attribute__((ext_vector_type(8)))  float    v8f;

#define S_LEN 2048
#define NH    16
#define HD    128
#define BQ    128   // query rows per workgroup
#define BK    64    // keys per tile
#define NWAVES 8

__global__ __launch_bounds__(256) void fa_fwd_gfx1250(const float* __restrict__ Q,
                                                      const float* __restrict__ K,
                                                      const float* __restrict__ V,
                                                      float* __restrict__ O)
{
    __shared__ __align__(32) _Float16 ldsK[BK * HD];           // [key][d]   f16
    __shared__ __align__(32) _Float16 ldsVt[HD * BK];          // [d][key]   f16 (transposed)
    __shared__ __align__(32) _Float16 ldsP[NWAVES][16 * BK];   // per-wave P [m][key]

    const int tid  = threadIdx.x;
    const int wave = tid >> 5;
    const int lane = tid & 31;
    const int lm   = lane & 15;   // column / M index inside 16-group
    const int hi   = lane >> 4;   // which half of the wave

    const int qbase = blockIdx.x * BQ;
    const int h     = blockIdx.y;
    const int b     = blockIdx.z;
    const int qw    = qbase + wave * 16;    // first query row of this wave
    const int qrow  = qw + lm;              // A-matrix row handled by this lane

    const float scale = 0.08838834764831845f;  // 1/sqrt(128)

    // ---- Load Q into WMMA A-fragments (16x32 f16 layout), pre-scaled ----
    // lane m=lm: halves [0..7]  = K  c*32 + hi*16     .. +7
    //            halves [8..15] = K  c*32 + hi*16 + 8 .. +7
    v16h qa[4];
    {
        const float* qptr = Q + ((size_t)(b * S_LEN + qrow) * NH + h) * HD;
        #pragma unroll
        for (int c = 0; c < 4; ++c) {
            const int d0 = c * 32 + hi * 16;
            const int d1 = d0 + 8;
            #pragma unroll
            for (int j = 0; j < 8; ++j) {
                qa[c][j]     = (_Float16)(qptr[d0 + j] * scale);
                qa[c][8 + j] = (_Float16)(qptr[d1 + j] * scale);
            }
        }
    }

    // ---- accumulators / online-softmax state ----
    v8f o[8];
    #pragma unroll
    for (int f = 0; f < 8; ++f) { v8f z = {}; o[f] = z; }
    float mrow[8], lrow[8];
    #pragma unroll
    for (int i = 0; i < 8; ++i) { mrow[i] = -3.0e38f; lrow[i] = 0.0f; }

    const int    ntiles = (qbase + BQ) / BK;                       // uniform per block
    const float* kbase  = K + ((size_t)b * S_LEN * NH + h) * HD;   // + s*NH*HD + d
    const float* vbase  = V + ((size_t)b * S_LEN * NH + h) * HD;

    for (int kt = 0; kt < ntiles; ++kt) {
        const int k0 = kt * BK;

        __syncthreads();   // previous iteration's LDS readers are done

        // ---- cooperative stage: K tile -> ldsK, V tile (transposed) -> ldsVt ----
        for (int idx = tid; idx < (BK * HD) / 4; idx += 256) {
            const int row = idx >> 5;            // key row in tile
            const int c4  = (idx & 31) * 4;      // d offset (multiple of 4)
            const size_t g = (size_t)(k0 + row) * (NH * HD) + c4;
            const float4 kv = *reinterpret_cast<const float4*>(kbase + g);
            const float4 vv = *reinterpret_cast<const float4*>(vbase + g);
            union { _Float16 hh[4]; uint2 u; } pk;
            pk.hh[0] = (_Float16)kv.x; pk.hh[1] = (_Float16)kv.y;
            pk.hh[2] = (_Float16)kv.z; pk.hh[3] = (_Float16)kv.w;
            *reinterpret_cast<uint2*>(&ldsK[row * HD + c4]) = pk.u;
            ldsVt[(c4 + 0) * BK + row] = (_Float16)vv.x;
            ldsVt[(c4 + 1) * BK + row] = (_Float16)vv.y;
            ldsVt[(c4 + 2) * BK + row] = (_Float16)vv.z;
            ldsVt[(c4 + 3) * BK + row] = (_Float16)vv.w;
        }
        __syncthreads();

        // ---- S = Q * K^T  (4 subtiles of 16 keys, each K-dim chunked by 32) ----
        v8f sc[4];
        #pragma unroll
        for (int t = 0; t < 4; ++t) {
            v8f acc = {};
            #pragma unroll
            for (int c = 0; c < 4; ++c) {
                // B fragment (32x16): lane column = key (t*16+lm),
                // holds features c*32 + hi*16 .. +15 (contiguous in ldsK row)
                const v16h bfrag = *reinterpret_cast<const v16h*>(
                    &ldsK[(t * 16 + lm) * HD + c * 32 + hi * 16]);
                acc = __builtin_amdgcn_wmma_f32_16x16x32_f16(
                    false, qa[c], false, bfrag, (short)0, acc, false, false);
            }
            // causal mask (unconditional: keeps EXEC all-1s around WMMA)
            const int kg = k0 + t * 16 + lm;
            #pragma unroll
            for (int i = 0; i < 8; ++i) {
                const int qg = qw + i + 8 * hi;
                if (kg > qg) acc[i] = -3.0e38f;
            }
            sc[t] = acc;
        }

        // ---- online softmax: row max over 64 keys ----
        float mnew[8], alpha[8];
        #pragma unroll
        for (int i = 0; i < 8; ++i) {
            float tm = fmaxf(fmaxf(sc[0][i], sc[1][i]), fmaxf(sc[2][i], sc[3][i]));
            tm = fmaxf(tm, __shfl_xor(tm, 1, 16));
            tm = fmaxf(tm, __shfl_xor(tm, 2, 16));
            tm = fmaxf(tm, __shfl_xor(tm, 4, 16));
            tm = fmaxf(tm, __shfl_xor(tm, 8, 16));
            mnew[i]  = fmaxf(mrow[i], tm);
            alpha[i] = __expf(mrow[i] - mnew[i]);
            mrow[i]  = mnew[i];
        }

        // ---- P = exp(S - m); stash in per-wave LDS for C-layout -> A-layout ----
        _Float16* pP = &ldsP[wave][0];
        #pragma unroll
        for (int t = 0; t < 4; ++t) {
            #pragma unroll
            for (int i = 0; i < 8; ++i) {
                const float s = sc[t][i];
                const float e = (s > -1.0e37f) ? __expf(s - mnew[i]) : 0.0f;
                sc[t][i] = e;
                pP[(i + 8 * hi) * BK + t * 16 + lm] = (_Float16)e;
            }
        }

        // ---- row sums + rescale running accumulators ----
        #pragma unroll
        for (int i = 0; i < 8; ++i) {
            float r = sc[0][i] + sc[1][i] + sc[2][i] + sc[3][i];
            r += __shfl_xor(r, 1, 16);
            r += __shfl_xor(r, 2, 16);
            r += __shfl_xor(r, 4, 16);
            r += __shfl_xor(r, 8, 16);
            lrow[i] = lrow[i] * alpha[i] + r;
        }
        #pragma unroll
        for (int f = 0; f < 8; ++f) {
            #pragma unroll
            for (int i = 0; i < 8; ++i) o[f][i] *= alpha[i];
        }

        // same-wave LDS RAW on pP: LDS is in-order per wave, wait to be explicit
        asm volatile("s_wait_dscnt 0" ::: "memory");

        // ---- O += P * V  (A = P 16x32 chunks, B = V^T fragments from ldsVt) ----
        #pragma unroll
        for (int kc = 0; kc < 2; ++kc) {
            const int pb = lm * BK + kc * 32 + hi * 16;
            const v8h plo = *reinterpret_cast<const v8h*>(&pP[pb]);
            const v8h phi = *reinterpret_cast<const v8h*>(&pP[pb + 8]);
            const v16h pa = __builtin_shufflevector(plo, phi,
                0, 1, 2, 3, 4, 5, 6, 7, 8, 9, 10, 11, 12, 13, 14, 15);
            #pragma unroll
            for (int f = 0; f < 8; ++f) {
                // B fragment: lane column = d (f*16+lm), keys kc*32 + hi*16 .. +15
                const v16h vb = *reinterpret_cast<const v16h*>(
                    &ldsVt[(f * 16 + lm) * BK + kc * 32 + hi * 16]);
                o[f] = __builtin_amdgcn_wmma_f32_16x16x32_f16(
                    false, pa, false, vb, (short)0, o[f], false, false);
            }
        }
    }

    // ---- normalize and write out (fp32, layout [B,S,H,D]) ----
    float* obase = O + ((size_t)(b * S_LEN + qw) * NH + h) * HD;
    #pragma unroll
    for (int i = 0; i < 8; ++i) {
        const int   m   = i + 8 * hi;
        const float inv = 1.0f / lrow[i];
        #pragma unroll
        for (int f = 0; f < 8; ++f) {
            obase[(size_t)m * (NH * HD) + f * 16 + lm] = o[f][i] * inv;
        }
    }
}

extern "C" void kernel_launch(void* const* d_in, const int* in_sizes, int n_in,
                              void* d_out, int out_size, void* d_ws, size_t ws_size,
                              hipStream_t stream) {
    (void)n_in; (void)out_size; (void)d_ws; (void)ws_size;
    const float* q = (const float*)d_in[0];
    const float* k = (const float*)d_in[1];
    const float* v = (const float*)d_in[2];
    float*       o = (float*)d_out;
    int B = in_sizes[0] / (S_LEN * NH * HD);
    if (B < 1) B = 1;
    dim3 grid(S_LEN / BQ, NH, B);
    fa_fwd_gfx1250<<<grid, dim3(256), 0, stream>>>(q, k, v, o);
}
